// GRASPUniqueGNET_6176162972357
// MI455X (gfx1250) — matrix operand order repaired and checked by reading, not compiled, not run on hardware
//
#include <hip/hip_runtime.h>

// ---------- types for WMMA ----------
typedef __attribute__((ext_vector_type(16))) __bf16 v16bf;
typedef __attribute__((ext_vector_type(8)))  float  v8f;

union FragBF { v16bf v; uint4 q[2]; };

#if __has_builtin(__builtin_amdgcn_global_load_async_to_lds_b128)
#define ASYNC_LDS_OK 1
typedef __attribute__((ext_vector_type(4))) int ivec4_t;
typedef __attribute__((address_space(1))) ivec4_t g4i_t;   // global int4
typedef __attribute__((address_space(3))) ivec4_t l4i_t;   // LDS int4
#endif

__device__ __forceinline__ unsigned short f2bf(float f) {
  unsigned int u = __float_as_uint(f);
  unsigned int r = u + 0x7fffu + ((u >> 16) & 1u);   // round-to-nearest-even
  return (unsigned short)(r >> 16);
}

// ---------- small utility kernels ----------
__global__ void fill_zero_kernel(float* p, int n) {
  int i = blockIdx.x * blockDim.x + threadIdx.x;
  if (i < n) p[i] = 0.0f;
}

__global__ void exp_kernel(const float* __restrict__ src, float* __restrict__ dst, int n) {
  int i = blockIdx.x * blockDim.x + threadIdx.x;
  if (i < n) dst[i] = expf(src[i]);
}

// x_norm = x / rowsum * 10000, converted to bf16 with K padded/zero-filled
__global__ __launch_bounds__(256) void rownorm_kernel(
    const float* __restrict__ x, unsigned short* __restrict__ out,
    int G, int Gp, float target) {
  __shared__ float red[256];
  int row = blockIdx.x, t = threadIdx.x;
  const float* xr = x + (size_t)row * G;
  float s = 0.f;
  for (int g = t; g < G; g += 256) s += xr[g];
  red[t] = s; __syncthreads();
  for (int off = 128; off > 0; off >>= 1) {
    if (t < off) red[t] += red[t + off];
    __syncthreads();
  }
  float scale = target / red[0];
  unsigned short* orow = out + (size_t)row * Gp;
  for (int g = t; g < Gp; g += 256)
    orow[g] = f2bf(g < G ? xr[g] * scale : 0.0f);
}

// convert f32 weight [M,K] row-major -> bf16 [Mp, ld] at column offset koff, *scale, zero-padded
__global__ void convw_kernel(const float* __restrict__ src, unsigned short* __restrict__ dst,
                             int M, int K, int Mp, int Ks, int ld, int koff, float scale) {
  int tid = blockIdx.x * blockDim.x + threadIdx.x;
  int total = Mp * Ks;
  if (tid >= total) return;
  int m = tid / Ks, k = tid % Ks;
  float v = (m < M && k < K) ? src[(size_t)m * K + k] * scale : 0.0f;
  dst[(size_t)m * ld + koff + k] = f2bf(v);
}

// convert dense f32 [N,C] -> bf16 at [n*ld + coff + c]
__global__ void conv_strided_kernel(const float* __restrict__ src, unsigned short* __restrict__ dst,
                                    int N, int C, int ld, int coff) {
  int tid = blockIdx.x * blockDim.x + threadIdx.x;
  if (tid >= N * C) return;
  int n = tid / C, c = tid % C;
  dst[(size_t)n * ld + coff + c] = f2bf(src[tid]);
}

// per-column sum & sum-of-squares with row-chunked atomics; stats = [sum(F), sumsq(F)]
__global__ __launch_bounds__(128) void col_stats_kernel(
    const float* __restrict__ x, int N, int F, int rows_per_chunk, float* __restrict__ stats) {
  int col = blockIdx.x * 128 + threadIdx.x;
  if (col >= F) return;
  int r0 = blockIdx.y * rows_per_chunk;
  int r1 = min(N, r0 + rows_per_chunk);
  float s = 0.f, ss = 0.f;
  for (int r = r0; r < r1; ++r) {
    float v = x[(size_t)r * F + col];
    s += v; ss += v * v;
  }
  atomicAdd(&stats[col], s);
  atomicAdd(&stats[F + col], ss);
}

// BatchNorm (training stats) + ReLU; optional f32 out, optional bf16 out at column offset
__global__ void bn_relu_kernel(const float* __restrict__ x, const float* __restrict__ stats,
                               const float* __restrict__ gamma, const float* __restrict__ beta,
                               int N, int F, float* __restrict__ f32out,
                               unsigned short* __restrict__ bfout, int ldbf, int coff) {
  int tid = blockIdx.x * blockDim.x + threadIdx.x;
  if (tid >= N * F) return;
  int n = tid / F, f = tid % F;
  float invN = 1.0f / (float)N;
  float mean = stats[f] * invN;
  float var  = stats[F + f] * invN - mean * mean;
  float inv  = rsqrtf(var + 1e-5f);
  float y = (x[tid] - mean) * inv * gamma[f] + beta[f];
  y = y > 0.f ? y : 0.f;
  if (f32out) f32out[tid] = y;
  if (bfout)  bfout[(size_t)n * ldbf + coff + f] = f2bf(y);
}

// edge aggregation: agg[dst] += z[src], cnt[dst] += 1  (32 threads per edge, 4 floats each)
__global__ void aggregate_kernel(const long long* __restrict__ ei, const float* __restrict__ z,
                                 float* __restrict__ agg, float* __restrict__ cnt, int E, int D) {
  int tid = blockIdx.x * blockDim.x + threadIdx.x;
  int e = tid >> 5, g = tid & 31;
  if (e >= E || g * 4 >= D) return;
  long long s = ei[e], d = ei[(size_t)E + e];
  const float4 zv = *(const float4*)(z + (size_t)s * D + g * 4);
  float* ap = agg + (size_t)d * D + g * 4;
  atomicAdd(ap + 0, zv.x);
  atomicAdd(ap + 1, zv.y);
  atomicAdd(ap + 2, zv.z);
  atomicAdd(ap + 3, zv.w);
  if (g == 0) atomicAdd(&cnt[d], 1.0f);
}

// agg / max(cnt,1) -> bf16 into A3 (K3-stride) at column offset 0
__global__ void norm_agg_kernel(const float* __restrict__ agg, const float* __restrict__ cnt,
                                unsigned short* __restrict__ a3, int N, int D, int ld) {
  int tid = blockIdx.x * blockDim.x + threadIdx.x;
  if (tid >= N * D) return;
  int n = tid / D, f = tid % D;
  float c = cnt[n]; c = c < 1.0f ? 1.0f : c;
  a3[(size_t)n * ld + f] = f2bf(agg[tid] / c);
}

// ---------- WMMA bf16 GEMM with LDS-staged, double-buffered weight tiles ----------
// Block = 256 threads = 8 waves; block tile = 128 rows x 64 features; Kp % 64 == 0.
// EPI 0: linear+bias | EPI 1: exp(linear+bias) | EPI 2: relu(linear+bias) + bf16 mirror
#define BSTRIDE 72   // padded LDS row stride (elems) to reduce bank conflicts
#define BBUF    (64 * BSTRIDE)

template <int EPI>
__global__ __launch_bounds__(256) void gemm_bf16_kernel(
    const unsigned short* __restrict__ A, int lda,
    const unsigned short* __restrict__ W, int ldw,
    const float* __restrict__ bias,
    float* __restrict__ out, int ldo, int Fout,
    unsigned short* __restrict__ bfout, int ldbf, int bfoff,
    int Rtiles, int Ftiles64, int Kp) {
  __shared__ __align__(16) unsigned short Bs[2][BBUF];

  int t = threadIdx.x;
  int lane = t & 31, wv = t >> 5;
  int rb = blockIdx.x / Ftiles64, fg = blockIdx.x % Ftiles64;
  int rt = rb * 8 + wv;
  // rt is wave-uniform; force the predicate into an SGPR so the guard becomes a
  // scalar branch (WMMA requires EXEC all-ones; avoid saveexec masking).
  bool active = (__builtin_amdgcn_readfirstlane(rt) < Rtiles);
  int half = lane >> 4, l16 = lane & 15;

  // staging role: thread t loads 32B of W (feature sf, k-quarter sk)
  int sf = t >> 2;            // 0..63
  int sk = (t & 3) * 16;      // 0,16,32,48
  const unsigned short* Wrow = W + (size_t)(fg * 64 + sf) * ldw + sk;
  unsigned short* sdst = &Bs[0][0] + sf * BSTRIDE + sk;

  const unsigned short* Abase =
      A + (size_t)((active ? rt : 0) * 16 + l16) * lda;

  v8f acc[4] = {};
  int nc = Kp >> 6;

  // ---- stage chunk 0 ----
  {
    const unsigned short* gp = Wrow;
    unsigned short* lp = sdst;
#ifdef ASYNC_LDS_OK
    __builtin_amdgcn_global_load_async_to_lds_b128(
        (g4i_t*)(unsigned long long)gp, (l4i_t*)(unsigned int)(size_t)lp, 0, 0);
    __builtin_amdgcn_global_load_async_to_lds_b128(
        (g4i_t*)(unsigned long long)(gp + 8), (l4i_t*)(unsigned int)(size_t)(lp + 8), 0, 0);
#else
    uint4 t0 = *(const uint4*)gp;
    uint4 t1 = *(const uint4*)(gp + 8);
    *(uint4*)lp = t0;
    *(uint4*)(lp + 8) = t1;
#endif
  }
  FragBF aC[2], aN[2];
  aC[0].q[0] = *(const uint4*)(Abase + 8 * half);
  aC[0].q[1] = *(const uint4*)(Abase + 16 + 8 * half);
  aC[1].q[0] = *(const uint4*)(Abase + 32 + 8 * half);
  aC[1].q[1] = *(const uint4*)(Abase + 48 + 8 * half);
#ifdef ASYNC_LDS_OK
#if __has_builtin(__builtin_amdgcn_s_wait_asynccnt)
  __builtin_amdgcn_s_wait_asynccnt(0);
#else
  asm volatile("s_wait_asynccnt 0x0" ::: "memory");
#endif
#endif
  __syncthreads();

  for (int c = 0; c < nc; ++c) {
    int cur = c & 1;

    // ---- preload ALL B fragments of this chunk into registers first:
    // 16 ds_load_b128 issue as a clause; latency overlaps the stage below,
    // and only one s_wait_dscnt lands before the first WMMA.
    FragBF b[2][4];
    if (active) {
#pragma unroll
      for (int s = 0; s < 2; ++s) {
        const unsigned short* bbase = &Bs[cur][0] + s * 32 + 8 * half;
#pragma unroll
        for (int j = 0; j < 4; ++j) {
          const unsigned short* lp2 = bbase + (j * 16 + l16) * BSTRIDE;
          b[s][j].q[0] = *(const uint4*)lp2;
          b[s][j].q[1] = *(const uint4*)(lp2 + 16);
        }
      }
    }

    if (c + 1 < nc) {
      // stage next weight chunk into the other LDS buffer
      const unsigned short* gp = Wrow + ((c + 1) << 6);
      unsigned short* lp = sdst + (1 - cur) * BBUF;
#ifdef ASYNC_LDS_OK
      __builtin_amdgcn_global_load_async_to_lds_b128(
          (g4i_t*)(unsigned long long)gp, (l4i_t*)(unsigned int)(size_t)lp, 0, 0);
      __builtin_amdgcn_global_load_async_to_lds_b128(
          (g4i_t*)(unsigned long long)(gp + 8), (l4i_t*)(unsigned int)(size_t)(lp + 8), 0, 0);
#else
      uint4 t0 = *(const uint4*)gp;
      uint4 t1 = *(const uint4*)(gp + 8);
      *(uint4*)lp = t0;
      *(uint4*)(lp + 8) = t1;
#endif
      // register double-buffer next A fragments
      const unsigned short* an = Abase + ((c + 1) << 6);
      aN[0].q[0] = *(const uint4*)(an + 8 * half);
      aN[0].q[1] = *(const uint4*)(an + 16 + 8 * half);
      aN[1].q[0] = *(const uint4*)(an + 32 + 8 * half);
      aN[1].q[1] = *(const uint4*)(an + 48 + 8 * half);
    }

    if (active) {
#pragma unroll
      for (int s = 0; s < 2; ++s) {
#pragma unroll
        for (int j = 0; j < 4; ++j) {
          acc[j] = __builtin_amdgcn_wmma_f32_16x16x32_bf16(
              false, aC[s].v, false, b[s][j].v, (short)0, acc[j], false, false);
        }
      }
    }

#ifdef ASYNC_LDS_OK
    if (c + 1 < nc) {
#if __has_builtin(__builtin_amdgcn_s_wait_asynccnt)
      __builtin_amdgcn_s_wait_asynccnt(0);
#else
      asm volatile("s_wait_asynccnt 0x0" ::: "memory");
#endif
    }
#endif
    __syncthreads();
    aC[0] = aN[0];
    aC[1] = aN[1];
  }

  if (!active) return;
#pragma unroll
  for (int j = 0; j < 4; ++j) {
    int col = fg * 64 + j * 16 + l16;
    if (col >= Fout) continue;
    float bv = bias ? bias[col] : 0.0f;
#pragma unroll
    for (int r = 0; r < 8; ++r) {
      int row = rt * 16 + r + half * 8;
      float v = acc[j][r] + bv;
      if (EPI == 1) v = expf(v);
      if (EPI == 2) v = v > 0.f ? v : 0.f;
      out[(size_t)row * ldo + col] = v;
      if (EPI == 2 && bfout)
        bfout[(size_t)row * ldbf + bfoff + col] = f2bf(v);
    }
  }
}

// ---------- host launcher ----------
static inline int cdiv_i(int a, int b) { return (a + b - 1) / b; }
static inline int rup_i(int a, int b) { return cdiv_i(a, b) * b; }

extern "C" void kernel_launch(void* const* d_in, const int* in_sizes, int n_in,
                              void* d_out, int out_size, void* d_ws, size_t ws_size,
                              hipStream_t stream) {
  const float* x_c1 = (const float*)d_in[0];
  const float* x_zc = (const float*)d_in[1];
  const long long* ei = (const long long*)d_in[2];
  const float* W1 = (const float*)d_in[3];
  const float* b1 = (const float*)d_in[4];
  const float* g1 = (const float*)d_in[5];
  const float* be1 = (const float*)d_in[6];
  const float* W2 = (const float*)d_in[7];
  const float* b2 = (const float*)d_in[8];
  const float* g2 = (const float*)d_in[9];
  const float* be2 = (const float*)d_in[10];
  const float* Wl = (const float*)d_in[11];
  const float* bl = (const float*)d_in[12];
  const float* Wr = (const float*)d_in[13];
  const float* Wc = (const float*)d_in[14];
  const float* Ws = (const float*)d_in[15];
  const float* bs = (const float*)d_in[16];
  const float* Wd = (const float*)d_in[17];
  const float* bd = (const float*)d_in[18];
  const float* disp = (const float*)d_in[19];
  const float* Wb = (const float*)d_in[20];
  const float* bb = (const float*)d_in[21];

  const int G  = in_sizes[19];          // 2000
  const int H1 = in_sizes[4];           // 512
  const int D  = in_sizes[12];          // 128
  const int C  = in_sizes[14] / D;      // 128
  const int NB = in_sizes[21];          // 4
  const int N  = in_sizes[0] / G;       // 50000 (multiple of 16)
  const int E  = in_sizes[2] / 2;       // 800000

  const int Gp   = rup_i(G, 64);        // K-padded G (2048), chunks of 64
  const int Gp64 = rup_i(G, 64);        // F-padded G (2048)
  const int K3   = 2 * D + C;           // 384
  const int CD   = C + D;               // 256
  const int NBp  = rup_i(NB, 64);       // 64
  const int Rtiles = N / 16;
  const int Rblk = cdiv_i(Rtiles, 8);   // row-blocks of 8 wave tiles

  // workspace bump allocator
  char* wsb = (char*)d_ws; size_t off = 0;
  auto alloc = [&](size_t bytes) -> void* {
    void* p = wsb + off; off = (off + bytes + 255) & ~(size_t)255; return p;
  };
  unsigned short* xnbf = (unsigned short*)alloc((size_t)N * Gp * 2);
  unsigned short* W1bf = (unsigned short*)alloc((size_t)H1 * Gp * 2);
  float*          hpre = (float*)alloc((size_t)N * H1 * 4);
  float*          st1  = (float*)alloc((size_t)2 * H1 * 4);
  unsigned short* hbf  = (unsigned short*)alloc((size_t)N * H1 * 2);
  unsigned short* W2bf = (unsigned short*)alloc((size_t)D * H1 * 2);
  float*          zpre = (float*)alloc((size_t)N * D * 4);
  float*          st2  = (float*)alloc((size_t)2 * D * 4);
  float*          zf   = (float*)alloc((size_t)N * D * 4);
  unsigned short* A3   = (unsigned short*)alloc((size_t)N * K3 * 2);
  float*          agg  = (float*)alloc((size_t)N * D * 4);
  float*          cnt  = (float*)alloc((size_t)N * 4);
  unsigned short* W3bf = (unsigned short*)alloc((size_t)D * K3 * 2);
  unsigned short* h2bf = (unsigned short*)alloc((size_t)N * CD * 2);
  unsigned short* Wsbf = (unsigned short*)alloc((size_t)Gp64 * CD * 2);
  unsigned short* Wdbf = (unsigned short*)alloc((size_t)Gp64 * CD * 2);
  unsigned short* Wbbf = (unsigned short*)alloc((size_t)NBp * D * 2);

  float* out = (float*)d_out;
  size_t o1 = (size_t)N * D;            // px_scale
  size_t o2 = o1 + (size_t)N * G;       // px_rate
  size_t o3 = o2 + (size_t)G;           // px_dropout
  size_t o4 = o3 + (size_t)N * G;       // batch_pred

  // --- zero-init accumulated buffers ---
  fill_zero_kernel<<<cdiv_i(2 * H1, 256), 256, 0, stream>>>(st1, 2 * H1);
  fill_zero_kernel<<<cdiv_i(2 * D, 256), 256, 0, stream>>>(st2, 2 * D);
  fill_zero_kernel<<<cdiv_i(N * D, 256), 256, 0, stream>>>(agg, N * D);
  fill_zero_kernel<<<cdiv_i(N, 256), 256, 0, stream>>>(cnt, N);

  // --- library-size normalization -> bf16 (padded K) ---
  rownorm_kernel<<<N, 256, 0, stream>>>(x_c1, xnbf, G, Gp, 10000.0f);

  // --- weight conversions (zero-padded, Wc folded with -1) ---
  convw_kernel<<<cdiv_i(H1 * Gp, 256), 256, 0, stream>>>(W1, W1bf, H1, G, H1, Gp, Gp, 0, 1.f);
  convw_kernel<<<cdiv_i(D * H1, 256), 256, 0, stream>>>(W2, W2bf, D, H1, D, H1, H1, 0, 1.f);
  convw_kernel<<<cdiv_i(D * D, 256), 256, 0, stream>>>(Wl, W3bf, D, D, D, D, K3, 0, 1.f);
  convw_kernel<<<cdiv_i(D * D, 256), 256, 0, stream>>>(Wr, W3bf, D, D, D, D, K3, D, 1.f);
  convw_kernel<<<cdiv_i(D * C, 256), 256, 0, stream>>>(Wc, W3bf, D, C, D, C, K3, 2 * D, -1.f);
  convw_kernel<<<cdiv_i(Gp64 * CD, 256), 256, 0, stream>>>(Ws, Wsbf, G, CD, Gp64, CD, CD, 0, 1.f);
  convw_kernel<<<cdiv_i(Gp64 * CD, 256), 256, 0, stream>>>(Wd, Wdbf, G, CD, Gp64, CD, CD, 0, 1.f);
  convw_kernel<<<cdiv_i(NBp * D, 256), 256, 0, stream>>>(Wb, Wbbf, NB, D, NBp, D, D, 0, 1.f);

  // --- x_zcommon -> bf16 into A3 (cols 2D..) and h2 (cols 0..C) ---
  conv_strided_kernel<<<cdiv_i(N * C, 256), 256, 0, stream>>>(x_zc, A3, N, C, K3, 2 * D);
  conv_strided_kernel<<<cdiv_i(N * C, 256), 256, 0, stream>>>(x_zc, h2bf, N, C, CD, 0);

  // --- encoder GEMM1: hpre = xnorm @ W1^T + b1 ---
  gemm_bf16_kernel<0><<<Rblk * (H1 / 64), 256, 0, stream>>>(
      xnbf, Gp, W1bf, Gp, b1, hpre, H1, H1, nullptr, 0, 0, Rtiles, H1 / 64, Gp);

  // --- BN1 stats + apply -> hbf ---
  {
    int chunks = 64, rpc = cdiv_i(N, 64);
    col_stats_kernel<<<dim3(cdiv_i(H1, 128), chunks), 128, 0, stream>>>(hpre, N, H1, rpc, st1);
  }
  bn_relu_kernel<<<cdiv_i(N * H1, 256), 256, 0, stream>>>(
      hpre, st1, g1, be1, N, H1, nullptr, hbf, H1, 0);

  // --- encoder GEMM2: zpre = h @ W2^T + b2 ---
  gemm_bf16_kernel<0><<<Rblk * (D / 64), 256, 0, stream>>>(
      hbf, H1, W2bf, H1, b2, zpre, D, D, nullptr, 0, 0, Rtiles, D / 64, H1);

  // --- BN2 stats + apply -> z (f32 for gather) and A3 cols D..2D ---
  {
    int chunks = 64, rpc = cdiv_i(N, 64);
    col_stats_kernel<<<dim3(cdiv_i(D, 128), chunks), 128, 0, stream>>>(zpre, N, D, rpc, st2);
  }
  bn_relu_kernel<<<cdiv_i(N * D, 256), 256, 0, stream>>>(
      zpre, st2, g2, be2, N, D, zf, A3, K3, D);

  // --- SAGE mean aggregation ---
  aggregate_kernel<<<cdiv_i(E * 32, 256), 256, 0, stream>>>(ei, zf, agg, cnt, E, D);
  norm_agg_kernel<<<cdiv_i(N * D, 256), 256, 0, stream>>>(agg, cnt, A3, N, D, K3);

  // --- fused z_unique = relu([agg|z|x_zc] @ [Wl|Wr|-Wc]^T + bl); mirror bf16 into h2 ---
  gemm_bf16_kernel<2><<<Rblk * (D / 64), 256, 0, stream>>>(
      A3, K3, W3bf, K3, bl, out, D, D, h2bf, CD, C, Rtiles, D / 64, K3);

  // --- px_rate = exp(disp) ---
  exp_kernel<<<cdiv_i(G, 256), 256, 0, stream>>>(disp, out + o2, G);

  // --- decoder: px_scale = exp(h2 @ Ws^T + bs), px_dropout = h2 @ Wd^T + bd ---
  gemm_bf16_kernel<1><<<Rblk * (Gp64 / 64), 256, 0, stream>>>(
      h2bf, CD, Wsbf, CD, bs, out + o1, G, G, nullptr, 0, 0, Rtiles, Gp64 / 64, CD);
  gemm_bf16_kernel<0><<<Rblk * (Gp64 / 64), 256, 0, stream>>>(
      h2bf, CD, Wdbf, CD, bd, out + o3, G, G, nullptr, 0, 0, Rtiles, Gp64 / 64, CD);

  // --- batch_pred = z_unique @ Wb^T + bb (A = z_unique bf16 slice of h2) ---
  gemm_bf16_kernel<0><<<Rblk * (NBp / 64), 256, 0, stream>>>(
      h2bf + C, CD, Wbbf, D, bb, out + o4, NB, NB, nullptr, 0, 0, Rtiles, NBp / 64, D);
}
